// FlashSelfAttention_38354057953636
// MI455X (gfx1250) — matrix operand order repaired and checked by reading, not compiled, not run on hardware
//
#include <hip/hip_runtime.h>
#include <stdint.h>

#define D_MODEL 1024
#define NHEAD   16
#define HD      64
#define TSEQ    2048
#define BATCH   2
#define MTOT    (BATCH * TSEQ)   // 4096

typedef __bf16 v16bf __attribute__((ext_vector_type(16)));
typedef float  v8f   __attribute__((ext_vector_type(8)));

union Frag {
    v16bf    v;
    uint4    q2[2];
    uint32_t u[8];
};

__device__ __forceinline__ uint16_t f2bf(float f) {
    union { float f; uint32_t u; } c; c.f = f;
    return (uint16_t)((c.u + 0x7FFFu + ((c.u >> 16) & 1u)) >> 16); // RNE
}

__device__ __forceinline__ v8f wmma_bf16(const Frag& a, const Frag& b, v8f c) {
    // D = A(16x32 bf16) * B(32x16 bf16) + C(16x16 f32)
    return __builtin_amdgcn_wmma_f32_16x16x32_bf16(
        false, a.v, false, b.v, (short)0, c, false, false);
}

// ---------------------------------------------------------------- converts
__global__ __launch_bounds__(256) void cvt_f32_to_bf16(
    const float4* __restrict__ in, ushort4* __restrict__ out, int n4) {
    int i = blockIdx.x * blockDim.x + threadIdx.x;
    if (i < n4) {
        float4 f = in[i];
        ushort4 o;
        o.x = f2bf(f.x); o.y = f2bf(f.y); o.z = f2bf(f.z); o.w = f2bf(f.w);
        out[i] = o;
    }
}

// ---------------------------------------------------------------- GEMM core
// 64x64 output block per wave: 4x4 tiles of 16x16, K=1024.
// A[M,1024], B[N,1024] bf16 row-major (K contiguous). 16 WMMAs per 16 b128 loads.
__device__ __forceinline__ void gemm_block_1024(const uint16_t* __restrict__ A,
                                                const uint16_t* __restrict__ B,
                                                int mbase, int nbase, int lane,
                                                v8f (&acc)[4][4]) {
    const int lm = lane & 15, half = lane >> 4;
    const uint16_t* arow[4];
    const uint16_t* brow[4];
#pragma unroll
    for (int i = 0; i < 4; ++i) {
        arow[i] = A + (size_t)(mbase + i * 16 + lm) * 1024;
        brow[i] = B + (size_t)(nbase + i * 16 + lm) * 1024;
        acc[i][0] = {}; acc[i][1] = {}; acc[i][2] = {}; acc[i][3] = {};
    }
    for (int kk = 0; kk < 1024; kk += 32) {
        Frag a[4], b[4];
#pragma unroll
        for (int mt = 0; mt < 4; ++mt) {
            // A 16x32 layout: half0 -> K = kk+0..7 (v0..3), kk+16..23 (v4..7); half1 -> +8
            a[mt].q2[0] = *(const uint4*)(arow[mt] + kk + half * 8);
            a[mt].q2[1] = *(const uint4*)(arow[mt] + kk + 16 + half * 8);
        }
#pragma unroll
        for (int nt = 0; nt < 4; ++nt) {
            // B 32x16 layout: half0 -> K = kk+0..7, kk+16..23; half1 -> +8
            b[nt].q2[0] = *(const uint4*)(brow[nt] + kk + half * 8);
            b[nt].q2[1] = *(const uint4*)(brow[nt] + kk + 16 + half * 8);
        }
        __builtin_prefetch(arow[0] + kk + 64, 0, 1);
#pragma unroll
        for (int mt = 0; mt < 4; ++mt)
#pragma unroll
            for (int nt = 0; nt < 4; ++nt)
                acc[mt][nt] = wmma_bf16(a[mt], b[nt], acc[mt][nt]);
    }
}

// ---------------------------------------------------------------- GEMM 1: QKV
// grid: 8 (M-block groups) x 48 (N blocks) = 384 blocks, 8 waves/block
__global__ __launch_bounds__(256) void gemm_qkv_kernel(
    const uint16_t* __restrict__ xb, const uint16_t* __restrict__ wib,
    uint16_t* __restrict__ qh, uint16_t* __restrict__ kh, uint16_t* __restrict__ vt) {
    const int wid = threadIdx.x >> 5, lane = threadIdx.x & 31;
    const int mbase = ((blockIdx.x & 7) * 8 + wid) * 64;  // 0..4032
    const int nbase = (blockIdx.x >> 3) * 64;             // 0..3008
    v8f acc[4][4];
    gemm_block_1024(xb, wib, mbase, nbase, lane, acc);

    const int lm = lane & 15, half = lane >> 4;
    // A 64-col N block lies wholly inside one section and one head.
    const int sect = nbase >> 10;        // 0=q 1=k 2=v
    const int h = (nbase & 1023) >> 6;
#pragma unroll
    for (int nt = 0; nt < 4; ++nt) {
        const int d = ((nbase + nt * 16) & 63) + lm;
#pragma unroll
        for (int mt = 0; mt < 4; ++mt) {
#pragma unroll
            for (int r = 0; r < 8; ++r) {
                const int m = mbase + mt * 16 + r + 8 * half;
                const int b = m >> 11, t = m & 2047;
                const uint16_t val = f2bf(acc[mt][nt][r]);
                const size_t bh = (size_t)(b * NHEAD + h);
                if (sect == 0)      qh[(bh * TSEQ + t) * HD + d] = val;
                else if (sect == 1) kh[(bh * TSEQ + t) * HD + d] = val;
                else                vt[(bh * HD + d) * TSEQ + t] = val;  // V^T
            }
        }
    }
}

// ---------------------------------------------------------------- attention
__global__ __launch_bounds__(256) void attn_kernel(
    const uint16_t* __restrict__ qh, const uint16_t* __restrict__ kh,
    const uint16_t* __restrict__ vt, uint16_t* __restrict__ ao) {
    __shared__ uint16_t probs[8][16 * 32];   // per-wave 16x32 bf16 tile

    const int wid = threadIdx.x >> 5, lane = threadIdx.x & 31;
    const int lm = lane & 15, half = lane >> 4;
    const int tile = blockIdx.x * 8 + wid;    // 0..4095
    const int bh = tile >> 7;                 // (b*16+h)
    const int qb = (tile & 127) << 4;         // query tile base
    const uint16_t* qp = qh + (size_t)bh * TSEQ * HD;
    const uint16_t* kp = kh + (size_t)bh * TSEQ * HD;
    const uint16_t* vp = vt + (size_t)bh * HD * TSEQ;

    // Q fragments: 16 queries x 64 d -> two 16x32 A-fragments, loaded once.
    Frag qf0, qf1;
    {
        const uint16_t* qrow = qp + (size_t)(qb + lm) * HD;
        qf0.q2[0] = *(const uint4*)(qrow + half * 8);
        qf0.q2[1] = *(const uint4*)(qrow + 16 + half * 8);
        qf1.q2[0] = *(const uint4*)(qrow + 32 + half * 8);
        qf1.q2[1] = *(const uint4*)(qrow + 48 + half * 8);
    }

    float mrun[8], lrun[8];
    v8f acc[4] = {{}, {}, {}, {}};
#pragma unroll
    for (int r = 0; r < 8; ++r) { mrun[r] = -3.0e38f; lrun[r] = 0.0f; }

    uint16_t* myp = &probs[wid][0];

    for (int kk = 0; kk < TSEQ; kk += 32) {
        // K fragments: two groups of 16 keys, each needs K=64 -> 2 B-fragments.
        Frag k00, k01, k10, k11;
        const uint16_t* krow0 = kp + (size_t)(kk + lm) * HD;
        const uint16_t* krow1 = kp + (size_t)(kk + 16 + lm) * HD;
        k00.q2[0] = *(const uint4*)(krow0 + half * 8);
        k00.q2[1] = *(const uint4*)(krow0 + 16 + half * 8);
        k01.q2[0] = *(const uint4*)(krow0 + 32 + half * 8);
        k01.q2[1] = *(const uint4*)(krow0 + 48 + half * 8);
        k10.q2[0] = *(const uint4*)(krow1 + half * 8);
        k10.q2[1] = *(const uint4*)(krow1 + 16 + half * 8);
        k11.q2[0] = *(const uint4*)(krow1 + 32 + half * 8);
        k11.q2[1] = *(const uint4*)(krow1 + 48 + half * 8);

        v8f s0 = {}, s1 = {};
        s0 = wmma_bf16(qf0, k00, s0);
        s0 = wmma_bf16(qf1, k01, s0);
        s1 = wmma_bf16(qf0, k10, s1);
        s1 = wmma_bf16(qf1, k11, s1);

        // C layout: row q = r + 8*half, col j = lane&15 (per 16-key group).
        const int j0 = kk + lm, j1 = kk + 16 + lm;
#pragma unroll
        for (int r = 0; r < 8; ++r) {
            const int qg = qb + r + 8 * half;
            // faithful reference: tril *zeroes* masked scores before softmax
            float a0 = (j0 <= qg) ? s0[r] * 0.125f : 0.0f;
            float a1 = (j1 <= qg) ? s1[r] * 0.125f : 0.0f;
            // per-row max over the 16 lanes of this half
            float cm = fmaxf(a0, a1);
            cm = fmaxf(cm, __shfl_xor(cm, 1, 32));
            cm = fmaxf(cm, __shfl_xor(cm, 2, 32));
            cm = fmaxf(cm, __shfl_xor(cm, 4, 32));
            cm = fmaxf(cm, __shfl_xor(cm, 8, 32));
            const float mnew  = fmaxf(mrun[r], cm);
            const float scale = __expf(mrun[r] - mnew);
            const float p0 = __expf(a0 - mnew);
            const float p1 = __expf(a1 - mnew);
            float rs = p0 + p1;
            rs += __shfl_xor(rs, 1, 32);
            rs += __shfl_xor(rs, 2, 32);
            rs += __shfl_xor(rs, 4, 32);
            rs += __shfl_xor(rs, 8, 32);
            lrun[r] = lrun[r] * scale + rs;
            mrun[r] = mnew;
            acc[0][r] *= scale; acc[1][r] *= scale;
            acc[2][r] *= scale; acc[3][r] *= scale;
            const int ql = r + 8 * half;
            myp[ql * 32 + lm]      = f2bf(p0);
            myp[ql * 32 + 16 + lm] = f2bf(p1);
        }

        // Reload probs from LDS in A-fragment layout (per-wave tile, DS in-order).
        Frag pf;
        const uint16_t* prow = myp + lm * 32;
        pf.q2[0] = *(const uint4*)(prow + half * 8);
        pf.q2[1] = *(const uint4*)(prow + 16 + half * 8);

        // out += P(16x32) * V^T-fragment (32 keys x 16 d), 4 d-tiles
#pragma unroll
        for (int dt = 0; dt < 4; ++dt) {
            Frag vf;
            const uint16_t* vrow = vp + (size_t)(dt * 16 + lm) * TSEQ + kk;
            vf.q2[0] = *(const uint4*)(vrow + half * 8);
            vf.q2[1] = *(const uint4*)(vrow + 16 + half * 8);
            acc[dt] = wmma_bf16(pf, vf, acc[dt]);
        }
    }

    // normalize and store bf16 into [b, t, h*64+d]
    const int b = bh >> 4, h = bh & 15;
#pragma unroll
    for (int dt = 0; dt < 4; ++dt) {
#pragma unroll
        for (int r = 0; r < 8; ++r) {
            const int qg = qb + r + 8 * half;
            const int d = dt * 16 + lm;
            ao[((size_t)(b * TSEQ + qg)) * D_MODEL + h * HD + d] =
                f2bf(acc[dt][r] / lrun[r]);
        }
    }
}

// ---------------------------------------------------------------- GEMM 2: proj
// grid: 8 (M-block groups) x 16 (N blocks) = 128 blocks
__global__ __launch_bounds__(256) void gemm_proj_kernel(
    const uint16_t* __restrict__ ao, const uint16_t* __restrict__ wob,
    float* __restrict__ out) {
    const int wid = threadIdx.x >> 5, lane = threadIdx.x & 31;
    const int mbase = ((blockIdx.x & 7) * 8 + wid) * 64;  // 0..4032
    const int nbase = (blockIdx.x >> 3) * 64;             // 0..960
    v8f acc[4][4];
    gemm_block_1024(ao, wob, mbase, nbase, lane, acc);

    const int lm = lane & 15, half = lane >> 4;
#pragma unroll
    for (int nt = 0; nt < 4; ++nt) {
        const int n = nbase + nt * 16 + lm;
#pragma unroll
        for (int mt = 0; mt < 4; ++mt) {
#pragma unroll
            for (int r = 0; r < 8; ++r) {
                const int m = mbase + mt * 16 + r + 8 * half;
                out[(size_t)m * D_MODEL + n] = acc[mt][nt][r];
            }
        }
    }
}

// ---------------------------------------------------------------- launch
extern "C" void kernel_launch(void* const* d_in, const int* in_sizes, int n_in,
                              void* d_out, int out_size, void* d_ws, size_t ws_size,
                              hipStream_t stream) {
    const float* x     = (const float*)d_in[0];
    const float* w_in  = (const float*)d_in[1];
    const float* w_out = (const float*)d_in[2];
    float* out = (float*)d_out;
    (void)in_sizes; (void)n_in; (void)out_size; (void)ws_size;

    char* ws = (char*)d_ws;
    size_t off = 0;
    auto carve = [&](size_t bytes) -> void* {
        void* p = ws + off;
        off += (bytes + 255) & ~(size_t)255;
        return p;
    };
    uint16_t* xb  = (uint16_t*)carve((size_t)MTOT * D_MODEL * 2);
    uint16_t* wib = (uint16_t*)carve((size_t)3 * D_MODEL * D_MODEL * 2);
    uint16_t* wob = (uint16_t*)carve((size_t)D_MODEL * D_MODEL * 2);
    uint16_t* qh  = (uint16_t*)carve((size_t)BATCH * NHEAD * TSEQ * HD * 2);
    uint16_t* kh  = (uint16_t*)carve((size_t)BATCH * NHEAD * TSEQ * HD * 2);
    uint16_t* vt  = (uint16_t*)carve((size_t)BATCH * NHEAD * TSEQ * HD * 2);
    uint16_t* ao  = (uint16_t*)carve((size_t)MTOT * D_MODEL * 2);

    int n4;
    n4 = MTOT * D_MODEL / 4;
    cvt_f32_to_bf16<<<(n4 + 255) / 256, 256, 0, stream>>>((const float4*)x, (ushort4*)xb, n4);
    n4 = 3 * D_MODEL * D_MODEL / 4;
    cvt_f32_to_bf16<<<(n4 + 255) / 256, 256, 0, stream>>>((const float4*)w_in, (ushort4*)wib, n4);
    n4 = D_MODEL * D_MODEL / 4;
    cvt_f32_to_bf16<<<(n4 + 255) / 256, 256, 0, stream>>>((const float4*)w_out, (ushort4*)wob, n4);

    // 64x48 blocks of 64x64, 8 waves/block along M
    gemm_qkv_kernel<<<8 * (3 * D_MODEL / 64), 256, 0, stream>>>(xb, wib, qh, kh, vt);
    // 4096 query tiles / 8 waves per block
    attn_kernel<<<(BATCH * NHEAD * (TSEQ / 16)) / 8, 256, 0, stream>>>(qh, kh, vt, ao);
    // 64x16 blocks of 64x64
    gemm_proj_kernel<<<8 * (D_MODEL / 64), 256, 0, stream>>>(ao, wob, out);
}
